// Attention_59725815218595
// MI455X (gfx1250) — matrix operand order repaired and checked by reading, not compiled, not run on hardware
//
#include <hip/hip_runtime.h>
#include <hip/hip_bf16.h>
#include <cstdint>

#define BATCH 2
#define SEQ   2048
#define DIMC  1024
#define HEADS 16
#define HD    64
#define ATTN_SCALE 0.125f   // 1/sqrt(64)

typedef __attribute__((ext_vector_type(16))) _Float16 v16h;
typedef __attribute__((ext_vector_type(8)))  _Float16 v8h;
typedef __attribute__((ext_vector_type(4)))  _Float16 v4h;
typedef __attribute__((ext_vector_type(8)))  float    v8f;

union V16HU { v16h v; v8h h[2]; };

// ---- async copy: global -> LDS (16B per lane), ASYNCcnt-tracked -------------
#if __has_builtin(__builtin_amdgcn_global_load_async_to_lds_b128)
#define HAS_ASYNC 1
#else
#define HAS_ASYNC 0
#endif

// builtin pointee is a gcc-vector int4 (confirmed by round-2 diagnostic)
typedef int v4i_b128 __attribute__((vector_size(16)));
typedef __attribute__((address_space(1))) v4i_b128* as1_b128_ptr;  // global
typedef __attribute__((address_space(3))) v4i_b128* as3_b128_ptr;  // LDS

static __device__ inline void cp16_g2lds(const _Float16* g, _Float16* l) {
#if HAS_ASYNC
  // generic->global AS cast is an identity; generic->LDS is low-32-bit truncate
  __builtin_amdgcn_global_load_async_to_lds_b128(
      (as1_b128_ptr)(uintptr_t)g,
      (as3_b128_ptr)(uint32_t)(uintptr_t)l,
      0, 0);
#else
  *(v8h*)l = *(const v8h*)g;    // global_load_b128 + ds_store_b128 fallback
#endif
}
#if HAS_ASYNC
#define ASYNC_WAIT_2() asm volatile("s_wait_asynccnt 0x2" ::: "memory")
#define ASYNC_WAIT_0() asm volatile("s_wait_asynccnt 0x0" ::: "memory")
#else
#define ASYNC_WAIT_2()
#define ASYNC_WAIT_0()
#endif

// ---- WMMA fragment loaders -------------------------------------------------
// A-matrix (16xK=32, f16): lane L -> row M=L%16.
// low lanes: elems 0-7 = K[0..8), elems 8-15 = K[16..24)
// hi  lanes: elems 0-7 = K[8..16), elems 8-15 = K[24..32)
static __device__ inline v16h load_a_f16(const _Float16* row, int hi) {
  V16HU u;
  u.h[0] = *(const v8h*)(row + hi * 8);
  u.h[1] = *(const v8h*)(row + hi * 8 + 16);
  return u.v;
}
static __device__ inline v16h load_a_f32(const float* row, int hi) {
  v16h a;
  const float* p0 = row + hi * 8;
  const float* p1 = p0 + 16;
#pragma unroll
  for (int i = 0; i < 8; ++i) { a[i] = (_Float16)p0[i]; a[8 + i] = (_Float16)p1[i]; }
  return a;
}
// B-matrix (K=32 x 16, f16): lane L -> column N=L%16.
// low lanes: elems 0-7 = K[0..8), 8-15 = K[16..24); hi lanes: K[8..16), K[24..32).
static __device__ inline v16h load_b_f16(const _Float16* colrow, int hi) {
  V16HU u;
  u.h[0] = *(const v8h*)(colrow + hi * 8);
  u.h[1] = *(const v8h*)(colrow + hi * 8 + 16);
  return u.v;
}

#define WMMA_F16(a, b, c) \
  __builtin_amdgcn_wmma_f32_16x16x32_f16(false, (a), false, (b), (short)0, (c), false, false)

// ---- cooperative B-tile stage: 32 cols x 32 k, f32 -> f16 into LDS ----------
// thread t: col = t>>3, k4 = (t&7)*4 -> one float4 load, 2 cvt_pk, 1 ds_store_b64
static __device__ inline void stage_b32x32(const float* wsrc, _Float16* sdst) {
  const float4 w4 = *(const float4*)wsrc;
  v4h hv;
  hv.x = (_Float16)w4.x; hv.y = (_Float16)w4.y;
  hv.z = (_Float16)w4.z; hv.w = (_Float16)w4.w;
  *(v4h*)sdst = hv;
}

// ---- Kernel 1: input projection  out = X @ W^T + bias  (f32 in, f16 out) ----
// MODE 0: split-head layout (b,h,n,d)   (for Q, K)
// MODE 1: transposed layout (b,h,d,n)   (for V, so PV B-operand is contiguous)
// Block tile 128x32: the 32x32 W strip is converted once into LDS and shared by
// all 8 waves; each wave computes a 16x32 tile (one A-fragment -> two WMMAs).
template <int MODE>
__global__ __launch_bounds__(256) void proj_gemm(
    const float* __restrict__ X, const float* __restrict__ W,
    const float* __restrict__ bias, _Float16* __restrict__ out) {
  __shared__ __align__(16) _Float16 sB[32 * 32];   // [col][k]

  const int tid  = threadIdx.x;
  const int lane = tid & 31;
  const int wave = tid >> 5;
  const int l16  = lane & 15;
  const int hi   = lane >> 4;
  const int mbase = blockIdx.y * 128 + wave * 16;
  const int nbase = blockIdx.x * 32;

  const float* arow = X + (size_t)(mbase + l16) * DIMC;
  const int scol = tid >> 3, sk = (tid & 7) * 4;
  const float* wsrc = W + (size_t)(nbase + scol) * DIMC + sk;
  _Float16* sdst = &sB[scol * 32 + sk];

  v8f acc0 = {}, acc1 = {};
  for (int kk = 0; kk < DIMC; kk += 32) {
    stage_b32x32(wsrc + kk, sdst);
    __syncthreads();
    v16h a  = load_a_f32(arow + kk, hi);
    v16h b0 = load_b_f16(sB + l16 * 32, hi);
    v16h b1 = load_b_f16(sB + (16 + l16) * 32, hi);
    acc0 = WMMA_F16(a, b0, acc0);
    acc1 = WMMA_F16(a, b1, acc1);
    __syncthreads();
  }

#pragma unroll
  for (int t = 0; t < 2; ++t) {
    const int c  = nbase + t * 16 + l16;
    const int h  = c >> 6, d = c & (HD - 1);
    const float bc = bias[c];
    const v8f& acc = t ? acc1 : acc0;
#pragma unroll
    for (int r = 0; r < 8; ++r) {
      const int m = mbase + r + hi * 8;
      const int bb = m >> 11;            // batch
      const int n  = m & (SEQ - 1);      // sequence pos
      const float v = acc[r] + bc;
      size_t idx;
      if (MODE == 0) idx = ((size_t)((bb * HEADS + h) * SEQ) + n) * HD + d;
      else           idx = ((size_t)((bb * HEADS + h) * HD)  + d) * SEQ + n;
      out[idx] = (_Float16)v;
    }
  }
}

// ---- Kernel 2: single-pass flash attention ---------------------------------
// One wave owns a 16-query tile; the 8 waves of a block share (b,h) and stage
// each 32-key K/V chunk into LDS via async global->LDS copies (double-buffered,
// ASYNCcnt + workgroup barriers). Scores computed transposed (S^T = K @ Q^T) so
// exp(S^T) in the C-layout IS the A-fragment for O = P @ V: no lane transposes.
__global__ __launch_bounds__(256) void flash_attn(
    const _Float16* __restrict__ qh, const _Float16* __restrict__ kh,
    const _Float16* __restrict__ vt, _Float16* __restrict__ ao) {
  __shared__ __align__(16) _Float16 sK[2][32 * HD];   // [key][d]
  __shared__ __align__(16) _Float16 sV[2][HD * 32];   // [d][key]

  const int tid  = threadIdx.x;
  const int lane = tid & 31;
  const int wave = tid >> 5;
  const int l16  = lane & 15;
  const int hi   = lane >> 4;
  const int bh   = blockIdx.y;
  const int b    = bh >> 4, h = bh & 15;
  const int qbase = (blockIdx.x * 8 + wave) * 16;

  const _Float16* qp = qh + (size_t)bh * SEQ * HD;
  const _Float16* kp = kh + (size_t)bh * SEQ * HD;
  const _Float16* vp = vt + (size_t)bh * HD * SEQ;

  // Q^T B-fragments (d is the K dim): lane column = query l16, contiguous d.
  v16h bq[2];
  bq[0] = load_b_f16(qp + (size_t)(qbase + l16) * HD, hi);
  bq[1] = load_b_f16(qp + (size_t)(qbase + l16) * HD + 32, hi);

  // cooperative stage of one 32-key chunk: K 4KB + V 4KB, 16B per thread each
  const int vr = tid >> 2, vs = tid & 3;  // V: 64 d-rows x 4 16B segments

  // prologue: chunk 0 -> buffer 0
  cp16_g2lds(kp + tid * 8, &sK[0][tid * 8]);
  cp16_g2lds(vp + (size_t)vr * SEQ + vs * 8, &sV[0][vr * 32 + vs * 8]);

  v8f O0 = {}, O1 = {}, O2 = {}, O3 = {};
  float m_run = -1e30f;
  float s_run = 0.0f;

  for (int jc = 0; jc < SEQ / 32; ++jc) {
    const int buf = jc & 1;
    if (jc + 1 < SEQ / 32) {          // prefetch next chunk into other buffer
      const int jn = (jc + 1) * 32;
      cp16_g2lds(kp + (size_t)jn * HD + tid * 8, &sK[buf ^ 1][tid * 8]);
      cp16_g2lds(vp + (size_t)vr * SEQ + jn + vs * 8, &sV[buf ^ 1][vr * 32 + vs * 8]);
      ASYNC_WAIT_2();                 // current chunk's 2 copies are done
    } else {
      ASYNC_WAIT_0();
    }
    __syncthreads();                  // all waves' portions visible

    const _Float16* Kb = sK[buf];
    const _Float16* Vb = sV[buf];

    // --- scores (transposed): t0 = keys 0-15 of chunk, t1 = keys 16-31 ---
    v8f t0 = {}, t1 = {};
#pragma unroll
    for (int s = 0; s < 2; ++s) {
      v16h ak0 = load_a_f16(Kb + l16 * HD + s * 32, hi);
      v16h ak1 = load_a_f16(Kb + (16 + l16) * HD + s * 32, hi);
      t0 = WMMA_F16(ak0, bq[s], t0);
      t1 = WMMA_F16(ak1, bq[s], t1);
    }

    // --- online softmax update (per-lane == per-query q = l16) ---
    float cm = t0[0];
#pragma unroll
    for (int i = 0; i < 8; ++i) { cm = fmaxf(cm, t0[i]); cm = fmaxf(cm, t1[i]); }
    cm = fmaxf(cm, __shfl_xor(cm, 16, 32));          // join the two key-halves
    const float m_new = fmaxf(m_run, cm);
    const float alpha = __expf(m_run - m_new);
    m_run = m_new;

    float psum = 0.0f;
    v16h ap;                                         // P as A-fragment, in place
#pragma unroll
    for (int i = 0; i < 8; ++i) {
      const float p0 = __expf(t0[i] - m_new);
      const float p1 = __expf(t1[i] - m_new);
      psum += p0 + p1;
      ap[i]     = (_Float16)p0;
      ap[8 + i] = (_Float16)p1;
    }
    s_run = s_run * alpha + psum;

    // --- V B-fragments from LDS (d-major rows of 32 keys) ---
    v16h bv0 = load_b_f16(Vb + ( 0 + l16) * 32, hi);
    v16h bv1 = load_b_f16(Vb + (16 + l16) * 32, hi);
    v16h bv2 = load_b_f16(Vb + (32 + l16) * 32, hi);
    v16h bv3 = load_b_f16(Vb + (48 + l16) * 32, hi);

    // rescale O rows: O-row q = qbase + r + 8*hi, stats live in lane (r + 8*hi)
#pragma unroll
    for (int r = 0; r < 8; ++r) {
      const float ar = __shfl(alpha, r + hi * 8, 16);
      O0[r] *= ar; O1[r] *= ar; O2[r] *= ar; O3[r] *= ar;
    }
    O0 = WMMA_F16(ap, bv0, O0);
    O1 = WMMA_F16(ap, bv1, O1);
    O2 = WMMA_F16(ap, bv2, O2);
    O3 = WMMA_F16(ap, bv3, O3);

    __syncthreads();                  // done reading buf before it is reloaded
  }

  const float s_tot = s_run + __shfl_xor(s_run, 16, 32);
  const float inv = ATTN_SCALE / s_tot;              // softmax * SCALE (post)

  _Float16* aop = ao + ((size_t)b * SEQ + qbase) * DIMC + h * HD;
#pragma unroll
  for (int r = 0; r < 8; ++r) {
    const float vvr = __shfl(inv, r + hi * 8, 16);
    const int row = r + hi * 8;
    aop[(size_t)row * DIMC +  0 + l16] = (_Float16)(O0[r] * vvr);
    aop[(size_t)row * DIMC + 16 + l16] = (_Float16)(O1[r] * vvr);
    aop[(size_t)row * DIMC + 32 + l16] = (_Float16)(O2[r] * vvr);
    aop[(size_t)row * DIMC + 48 + l16] = (_Float16)(O3[r] * vvr);
  }
}

// ---- Kernel 3: output projection  out = AO @ Wp^T + bp  (f16 A, f32 out) ----
__global__ __launch_bounds__(256) void out_proj(
    const _Float16* __restrict__ A, const float* __restrict__ W,
    const float* __restrict__ bias, float* __restrict__ out) {
  __shared__ __align__(16) _Float16 sB[32 * 32];   // [col][k]

  const int tid  = threadIdx.x;
  const int lane = tid & 31;
  const int wave = tid >> 5;
  const int l16  = lane & 15;
  const int hi   = lane >> 4;
  const int mbase = blockIdx.y * 128 + wave * 16;
  const int nbase = blockIdx.x * 32;

  const _Float16* arow = A + (size_t)(mbase + l16) * DIMC;
  const int scol = tid >> 3, sk = (tid & 7) * 4;
  const float* wsrc = W + (size_t)(nbase + scol) * DIMC + sk;
  _Float16* sdst = &sB[scol * 32 + sk];

  v8f acc0 = {}, acc1 = {};
  for (int kk = 0; kk < DIMC; kk += 32) {
    stage_b32x32(wsrc + kk, sdst);
    __syncthreads();
    v16h a  = load_a_f16(arow + kk, hi);
    v16h b0 = load_b_f16(sB + l16 * 32, hi);
    v16h b1 = load_b_f16(sB + (16 + l16) * 32, hi);
    acc0 = WMMA_F16(a, b0, acc0);
    acc1 = WMMA_F16(a, b1, acc1);
    __syncthreads();
  }

#pragma unroll
  for (int t = 0; t < 2; ++t) {
    const int c = nbase + t * 16 + l16;
    const float bc = bias[c];
    const v8f& acc = t ? acc1 : acc0;
#pragma unroll
    for (int r = 0; r < 8; ++r) {
      const int m = mbase + r + hi * 8;
      out[(size_t)m * DIMC + c] = acc[r] + bc;
    }
  }
}

// ---- Launch ----------------------------------------------------------------
extern "C" void kernel_launch(void* const* d_in, const int* in_sizes, int n_in,
                              void* d_out, int out_size, void* d_ws, size_t ws_size,
                              hipStream_t stream) {
  (void)in_sizes; (void)n_in; (void)out_size; (void)ws_size;
  const float* query = (const float*)d_in[0];
  const float* key   = (const float*)d_in[1];
  const float* value = (const float*)d_in[2];
  const float* Wq = (const float*)d_in[3];
  const float* bq = (const float*)d_in[4];
  const float* Wk = (const float*)d_in[5];
  const float* bk = (const float*)d_in[6];
  const float* Wv = (const float*)d_in[7];
  const float* bv = (const float*)d_in[8];
  const float* Wp = (const float*)d_in[9];
  const float* bp = (const float*)d_in[10];
  float* out = (float*)d_out;

  const size_t ELT = (size_t)BATCH * SEQ * DIMC;  // 4,194,304 halves per buffer
  _Float16* ws = (_Float16*)d_ws;
  _Float16* qh = ws;                 // (b,h,n,d) f16
  _Float16* kh = ws + ELT;           // (b,h,n,d) f16
  _Float16* vt = ws + 2 * ELT;       // (b,h,d,n) f16 (transposed)
  _Float16* ao = ws + 3 * ELT;       // (b,n,dim) f16 attention output

  dim3 blk(256, 1, 1);
  dim3 g1(DIMC / 32, (BATCH * SEQ) / 128, 1);   // 32 x 32 blocks, 8 waves each
  proj_gemm<0><<<g1, blk, 0, stream>>>(query, Wq, bq, qh);
  proj_gemm<0><<<g1, blk, 0, stream>>>(key,   Wk, bk, kh);
  proj_gemm<1><<<g1, blk, 0, stream>>>(value, Wv, bv, vt);

  dim3 g2(SEQ / 128, BATCH * HEADS, 1);         // 16 x 32 blocks, 8 q-tiles each
  flash_attn<<<g2, blk, 0, stream>>>(qh, kh, vt, ao);

  out_proj<<<g1, blk, 0, stream>>>(ao, Wp, bp, out);
}
